// MambaEncoderLayer_44074954392239
// MI455X (gfx1250) — matrix operand order repaired and checked by reading, not compiled, not run on hardware
//
#include <hip/hip_runtime.h>
#include <cstddef>
#include <cstdint>

// ---------------------------------------------------------------------------
// Types for CDNA5 WMMA (wave32): v_wmma_f32_16x16x32_bf16
// ---------------------------------------------------------------------------
typedef __bf16 bf16_t;
typedef __attribute__((ext_vector_type(16))) __bf16 v16bf;
typedef __attribute__((ext_vector_type(8)))  float  v8f;

#define D_MODEL 1024
#define D_IN    1024
#define N_STATE 16
#define DT_RANK 64
#define D_FFN   4096
#define BATCH   2
#define SEQ     2048
#define MROWS   (BATCH * SEQ)   // 4096 token rows
#define XDBL_LD 128             // x_dbl padded from 96 -> 128 columns

#define LDS_PAD 40              // padded row length (elements) for 32-wide K tiles

// ---------------------------------------------------------------------------
// CDNA5 async global->LDS copy (ASYNCcnt-tracked), 16 bytes per lane.
// Generic pointers to __shared__ carry the LDS byte offset in their low 32
// bits (flat->LDS truncation, ISA 10.2), which is what VDST wants.
// ---------------------------------------------------------------------------
__device__ __forceinline__ unsigned lds_addr32(const void* p) {
    return (unsigned)(unsigned long long)p;
}

__device__ __forceinline__ void async_copy_b128(unsigned lds_off, const void* gptr) {
    unsigned long long ga = (unsigned long long)gptr;
    asm volatile("global_load_async_to_lds_b128 %0, %1, off"
                 :: "v"(lds_off), "v"(ga)
                 : "memory");
}

__device__ __forceinline__ void wait_async_le3() {
    asm volatile("s_wait_asynccnt 0x3" ::: "memory");
}
__device__ __forceinline__ void wait_async_0() {
    asm volatile("s_wait_asynccnt 0x0" ::: "memory");
}

// ---------------------------------------------------------------------------
// fp32 -> bf16 weight conversion with optional zero row padding
// ---------------------------------------------------------------------------
__global__ __launch_bounds__(256)
void cvt_pad_kernel(const float* __restrict__ src, bf16_t* __restrict__ dst,
                    int rows_src, int total, int cols) {
    int i = blockIdx.x * blockDim.x + threadIdx.x;
    if (i >= total) return;
    int r = i / cols;
    dst[i] = (r < rows_src) ? (bf16_t)src[i] : (bf16_t)0.0f;
}

// ---------------------------------------------------------------------------
// LayerNorm over C columns per row. Optional: bf16 output of LN value
// (feeds next WMMA GEMM), fp32 output of (residual + LN value).
// ---------------------------------------------------------------------------
__global__ __launch_bounds__(256)
void ln_kernel(const float* __restrict__ x, const float* __restrict__ g,
               const float* __restrict__ b, bf16_t* __restrict__ outB,
               float* __restrict__ outF, const float* __restrict__ residual,
               int C) {
    __shared__ float s_sum[256];
    __shared__ float s_sq[256];
    const int row = blockIdx.x;
    const int tid = threadIdx.x;
    const size_t base = (size_t)row * C;

    float lsum = 0.f, lsq = 0.f;
    for (int c = tid; c < C; c += 256) {
        float v = x[base + c];
        lsum += v; lsq += v * v;
    }
    s_sum[tid] = lsum; s_sq[tid] = lsq;
    __syncthreads();
    for (int s = 128; s > 0; s >>= 1) {
        if (tid < s) { s_sum[tid] += s_sum[tid + s]; s_sq[tid] += s_sq[tid + s]; }
        __syncthreads();
    }
    const float mean = s_sum[0] / (float)C;
    const float var  = s_sq[0] / (float)C - mean * mean;
    const float inv  = rsqrtf(var + 1e-5f);

    for (int c = tid; c < C; c += 256) {
        float v = (x[base + c] - mean) * inv * g[c] + b[c];
        if (outB) outB[base + c] = (bf16_t)v;
        if (outF) outF[base + c] = v + (residual ? residual[base + c] : 0.0f);
    }
}

// ---------------------------------------------------------------------------
// WMMA fragment gathers from LDS tiles (row length LDS_PAD elements),
// per CDNA5 ISA 7.12.2 layouts (wave32). Each gather = 2x ds_load_b128.
// A: VGPR p holds k = (p<4?0:16) + half*8 + (p&3)*2 (+0,+1), row m = lane&15.
// B: VGPR p holds k = (p<4?0:16) + half*8 + (p&3)*2 (+0,+1), row n = lane&15.
// ---------------------------------------------------------------------------
__device__ __forceinline__ v16bf frag_A_lds(const bf16_t* sA, int mrel) {
    const int lane = threadIdx.x & 31;
    const int half = lane >> 4;
    const bf16_t* r = sA + (size_t)(mrel + (lane & 15)) * LDS_PAD;
    v16bf a;
#pragma unroll
    for (int p = 0; p < 8; ++p) {
        int kb = ((p & 3) * 2) + half * 8 + ((p >> 2) << 4);
        a[2 * p]     = r[kb];
        a[2 * p + 1] = r[kb + 1];
    }
    return a;
}

__device__ __forceinline__ v16bf frag_B_lds(const bf16_t* sB, int nrel) {
    const int lane = threadIdx.x & 31;
    const int half = lane >> 4;
    const bf16_t* r = sB + (size_t)(nrel + (lane & 15)) * LDS_PAD;
    v16bf b;
#pragma unroll
    for (int p = 0; p < 8; ++p) {
        int kb = ((p & 3) * 2) + half * 8 + ((p >> 2) << 4);
        b[2 * p]     = r[kb];
        b[2 * p + 1] = r[kb + 1];
    }
    return b;
}

// Issue async copies for one 64x32 A tile + 128x32 B tile (3 x 16B per thread)
__device__ __forceinline__ void issue_tile(const bf16_t* __restrict__ A, int lda,
                                           const bf16_t* __restrict__ W, int ldw,
                                           int m0, int n0, int k0,
                                           bf16_t* sA, bf16_t* sB) {
    const int t = threadIdx.x;
    {   // A: 64 rows x 32 cols = 256 chunks of 16B
        const int row = t >> 2, seg = t & 3;
        const bf16_t* g = A + (size_t)(m0 + row) * lda + k0 + seg * 8;
        async_copy_b128(lds_addr32(sA + row * LDS_PAD + seg * 8), g);
    }
#pragma unroll
    for (int c = 0; c < 2; ++c) {  // B: 128 rows x 32 cols = 512 chunks
        const int idx = t + c * 256;
        const int row = idx >> 2, seg = idx & 3;
        const bf16_t* g = W + (size_t)(n0 + row) * ldw + k0 + seg * 8;
        async_copy_b128(lds_addr32(sB + row * LDS_PAD + seg * 8), g);
    }
}

// ---------------------------------------------------------------------------
// GEMM: C[M,N] = act(A[M,K] @ W[N,K]^T + bias) (+ residual)
//   A bf16 (lda), W bf16 (ldw). 256 threads = 8 waves; block tile 64M x 128N;
//   each wave 32x32 (4 accumulators). LDS double-buffered via async copies.
//   Requires M%64==0, N%128==0, K%32==0 (buffers padded to guarantee this).
//   act: 0=none, 1=softplus (log1p(exp(x))), 2=LeakyReLU(0.01)
// ---------------------------------------------------------------------------
__global__ __launch_bounds__(256)
void gemm_bf16_kernel(const bf16_t* __restrict__ A, int lda,
                      const bf16_t* __restrict__ W, int ldw,
                      float* __restrict__ outF, bf16_t* __restrict__ outB,
                      int ldc, const float* __restrict__ bias,
                      const float* __restrict__ residual, int act,
                      int M, int N, int K) {
    (void)M; (void)N;
    __shared__ bf16_t sA[2][64 * LDS_PAD];
    __shared__ bf16_t sB[2][128 * LDS_PAD];

    const int lane = threadIdx.x & 31;
    const int wave = threadIdx.x >> 5;
    const int wm0 = (wave >> 2) * 32;     // 0,32
    const int wn0 = (wave & 3) * 32;      // 0..96
    const int m0 = blockIdx.y * 64;
    const int n0 = blockIdx.x * 128;
    const int half = lane >> 4;

    v8f c00 = {}, c01 = {}, c10 = {}, c11 = {};

    const int ntiles = K >> 5;
    issue_tile(A, lda, W, ldw, m0, n0, 0, sA[0], sB[0]);

    for (int i = 0; i < ntiles; ++i) {
        const int cur = i & 1;
        if (i + 1 < ntiles) {
            issue_tile(A, lda, W, ldw, m0, n0, (i + 1) << 5, sA[cur ^ 1], sB[cur ^ 1]);
            wait_async_le3();   // our 3 copies for tile i have landed
        } else {
            wait_async_0();
        }
        __syncthreads();        // all waves' copies for tile i visible

        v16bf a0 = frag_A_lds(sA[cur], wm0);
        v16bf a1 = frag_A_lds(sA[cur], wm0 + 16);
        v16bf b0 = frag_B_lds(sB[cur], wn0);
        v16bf b1 = frag_B_lds(sB[cur], wn0 + 16);
        c00 = __builtin_amdgcn_wmma_f32_16x16x32_bf16(false, a0, false, b0,
                                                      (short)0, c00, false, false);
        c01 = __builtin_amdgcn_wmma_f32_16x16x32_bf16(false, a0, false, b1,
                                                      (short)0, c01, false, false);
        c10 = __builtin_amdgcn_wmma_f32_16x16x32_bf16(false, a1, false, b0,
                                                      (short)0, c10, false, false);
        c11 = __builtin_amdgcn_wmma_f32_16x16x32_bf16(false, a1, false, b1,
                                                      (short)0, c11, false, false);
        __syncthreads();        // buffer `cur` free to be overwritten at i+2
    }

    // C/D layout: VGPR p -> m = p + 8*half; n = lane&15
    auto store_frag = [&](v8f c, int mbase, int nbase) {
        const int n = nbase + (lane & 15);
        const float bv = bias ? bias[n] : 0.0f;
#pragma unroll
        for (int p = 0; p < 8; ++p) {
            const int m = mbase + p + 8 * half;
            float v = c[p] + bv;
            if (act == 1)      v = (v > 20.0f) ? v : log1pf(__expf(v));
            else if (act == 2) v = (v >= 0.0f) ? v : 0.01f * v;
            const size_t idx = (size_t)m * ldc + n;
            if (outB) outB[idx] = (bf16_t)v;
            if (outF) outF[idx] = v + (residual ? residual[idx] : 0.0f);
        }
    };
    store_frag(c00, m0 + wm0,      n0 + wn0);
    store_frag(c01, m0 + wm0,      n0 + wn0 + 16);
    store_frag(c10, m0 + wm0 + 16, n0 + wn0);
    store_frag(c11, m0 + wm0 + 16, n0 + wn0 + 16);
}

// ---------------------------------------------------------------------------
// Depthwise causal conv1d (k=2) + SiLU; xz is (rows, 2*D_IN), x_in = cols 0..D_IN-1
// ---------------------------------------------------------------------------
__global__ __launch_bounds__(256)
void conv_silu_kernel(const float* __restrict__ xz, const float* __restrict__ cw,
                      const float* __restrict__ cb, float* __restrict__ xcF,
                      bf16_t* __restrict__ xcB, int Lseq, int total) {
    int i = blockIdx.x * blockDim.x + threadIdx.x;
    if (i >= total) return;
    const int d = i % D_IN;
    const int row = i / D_IN;
    const int l = row % Lseq;
    const float x  = xz[(size_t)row * (2 * D_IN) + d];
    const float xp = (l > 0) ? xz[(size_t)(row - 1) * (2 * D_IN) + d] : 0.0f;
    float v = xp * cw[2 * d] + x * cw[2 * d + 1] + cb[d];
    float s = v / (1.0f + __expf(-v));
    xcF[i] = s;
    xcB[i] = (bf16_t)s;
}

// ---------------------------------------------------------------------------
// Selective scan: one thread per (b, d) channel, 16 states in registers.
// x_dbl row layout (ld=128): [0..63]=dt, [64..79]=B, [80..95]=C.
// ---------------------------------------------------------------------------
__global__ __launch_bounds__(256)
void scan_kernel(const float* __restrict__ delta, const float* __restrict__ xdbl,
                 const float* __restrict__ xc, const float* __restrict__ A_log,
                 float* __restrict__ y, int Lseq) {
    const int gid = blockIdx.x * blockDim.x + threadIdx.x;   // 0 .. BATCH*D_IN-1
    const int b = gid / D_IN;
    const int d = gid % D_IN;

    float Arow[N_STATE];
#pragma unroll
    for (int n = 0; n < N_STATE; ++n) Arow[n] = -__expf(A_log[d * N_STATE + n]);

    float h[N_STATE];
#pragma unroll
    for (int n = 0; n < N_STATE; ++n) h[n] = 0.0f;

    for (int l = 0; l < Lseq; ++l) {
        const size_t row = (size_t)b * Lseq + l;
        const float dv = delta[row * D_IN + d];
        const float xv = xc[row * D_IN + d];
        const float* bd = xdbl + row * XDBL_LD;
        float yacc = 0.0f;
#pragma unroll
        for (int n = 0; n < N_STATE; ++n) {
            const float dA = __expf(dv * Arow[n]);
            h[n] = dA * h[n] + dv * bd[64 + n] * xv;
            yacc += h[n] * bd[80 + n];
        }
        y[row * D_IN + d] = yacc;
    }
}

// ---------------------------------------------------------------------------
// Gate: y2 = (y + xc*Dp) * silu(z), output bf16 for out_proj GEMM
// ---------------------------------------------------------------------------
__global__ __launch_bounds__(256)
void gate_kernel(const float* __restrict__ y, const float* __restrict__ xc,
                 const float* __restrict__ xz, const float* __restrict__ Dp,
                 bf16_t* __restrict__ y2, int total) {
    int i = blockIdx.x * blockDim.x + threadIdx.x;
    if (i >= total) return;
    const int d = i % D_IN;
    const int row = i / D_IN;
    const float z = xz[(size_t)row * (2 * D_IN) + D_IN + d];
    const float sz = z / (1.0f + __expf(-z));
    y2[i] = (bf16_t)((y[i] + xc[i] * Dp[d]) * sz);
}

// ---------------------------------------------------------------------------
extern "C" void kernel_launch(void* const* d_in, const int* in_sizes, int n_in,
                              void* d_out, int out_size, void* d_ws, size_t ws_size,
                              hipStream_t stream) {
    (void)in_sizes; (void)n_in; (void)out_size; (void)ws_size;
    const float* src        = (const float*)d_in[0];
    const float* in_proj_w  = (const float*)d_in[1];
    const float* conv_w     = (const float*)d_in[2];
    const float* conv_b     = (const float*)d_in[3];
    const float* x_proj_w   = (const float*)d_in[4];
    const float* dt_proj_w  = (const float*)d_in[5];
    const float* dt_proj_b  = (const float*)d_in[6];
    const float* A_log      = (const float*)d_in[7];
    const float* Dp         = (const float*)d_in[8];
    const float* out_proj_w = (const float*)d_in[9];
    const float* mnorm_g    = (const float*)d_in[10];
    const float* mnorm_b    = (const float*)d_in[11];
    const float* n1_g = (const float*)d_in[12]; const float* n1_b = (const float*)d_in[13];
    const float* n2_g = (const float*)d_in[14]; const float* n2_b = (const float*)d_in[15];
    const float* n3_g = (const float*)d_in[16]; const float* n3_b = (const float*)d_in[17];
    const float* ffn_w1 = (const float*)d_in[18];
    const float* ffn_w2 = (const float*)d_in[19];
    const float* fe_w   = (const float*)d_in[20];
    const float* fe_b   = (const float*)d_in[21];

    char* ws = (char*)d_ws;
    size_t off = 0;
    auto alloc = [&](size_t bytes) -> void* {
        void* p = ws + off;
        off += (bytes + 255) & ~(size_t)255;
        return p;
    };

    // bf16 weights
    bf16_t* wInB  = (bf16_t*)alloc((size_t)2 * D_IN * D_MODEL * 2);
    bf16_t* wXpB  = (bf16_t*)alloc((size_t)XDBL_LD * D_IN * 2);   // zero-padded 96->128 rows
    bf16_t* wDtB  = (bf16_t*)alloc((size_t)D_IN * DT_RANK * 2);
    bf16_t* wOutB = (bf16_t*)alloc((size_t)D_MODEL * D_IN * 2);
    bf16_t* wF1B  = (bf16_t*)alloc((size_t)D_FFN * D_MODEL * 2);
    bf16_t* wF2B  = (bf16_t*)alloc((size_t)D_MODEL * D_FFN * 2);
    bf16_t* wFeB  = (bf16_t*)alloc((size_t)D_MODEL * D_MODEL * 2);
    // activations (reused across phases)
    bf16_t* lnB   = (bf16_t*)alloc((size_t)MROWS * D_MODEL * 2);      // ln1 / ln2 / ln3
    float*  bufA  = (float*) alloc((size_t)MROWS * 2 * D_IN * 4);     // xz; later ffn hidden (bf16)
    float*  bufB  = (float*) alloc((size_t)MROWS * D_IN * 4);         // x_c f32; later mamba h
    float*  bufC  = (float*) alloc((size_t)MROWS * D_IN * 4);         // delta; later src3
    float*  bufD  = (float*) alloc((size_t)MROWS * D_IN * 4);         // y; later src2
    bf16_t* bufE  = (bf16_t*)alloc((size_t)MROWS * D_IN * 2);         // x_c bf16; later y2 bf16
    float*  xdblF = (float*) alloc((size_t)MROWS * XDBL_LD * 4);
    bf16_t* xdblB = (bf16_t*)alloc((size_t)MROWS * XDBL_LD * 2);

    const dim3 blk(256);
    auto cdiv = [](size_t a, size_t b) { return (unsigned)((a + b - 1) / b); };
    const size_t NTOK = (size_t)MROWS * D_IN;

    // --- weight conversions (fp32 -> bf16, x_proj zero-padded to 128 rows) ---
    cvt_pad_kernel<<<cdiv((size_t)2*D_IN*D_MODEL,256), blk, 0, stream>>>(in_proj_w, wInB, 2*D_IN, 2*D_IN*D_MODEL, D_MODEL);
    cvt_pad_kernel<<<cdiv((size_t)XDBL_LD*D_IN,256),   blk, 0, stream>>>(x_proj_w, wXpB, DT_RANK + 2*N_STATE, XDBL_LD*D_IN, D_IN);
    cvt_pad_kernel<<<cdiv((size_t)D_IN*DT_RANK,256),   blk, 0, stream>>>(dt_proj_w, wDtB, D_IN, D_IN*DT_RANK, DT_RANK);
    cvt_pad_kernel<<<cdiv((size_t)D_MODEL*D_IN,256),   blk, 0, stream>>>(out_proj_w, wOutB, D_MODEL, D_MODEL*D_IN, D_IN);
    cvt_pad_kernel<<<cdiv((size_t)D_FFN*D_MODEL,256),  blk, 0, stream>>>(ffn_w1, wF1B, D_FFN, D_FFN*D_MODEL, D_MODEL);
    cvt_pad_kernel<<<cdiv((size_t)D_MODEL*D_FFN,256),  blk, 0, stream>>>(ffn_w2, wF2B, D_MODEL, D_MODEL*D_FFN, D_FFN);
    cvt_pad_kernel<<<cdiv((size_t)D_MODEL*D_MODEL,256),blk, 0, stream>>>(fe_w, wFeB, D_MODEL, D_MODEL*D_MODEL, D_MODEL);

    // --- 1) ln1(src) -> bf16 ---
    ln_kernel<<<MROWS, blk, 0, stream>>>(src, n1_g, n1_b, lnB, nullptr, nullptr, D_MODEL);
    // --- 2) xz = ln1 @ in_proj^T (4096 x 2048) ---
    gemm_bf16_kernel<<<dim3((2*D_IN)/128, MROWS/64), blk, 0, stream>>>(
        lnB, D_MODEL, wInB, D_MODEL, bufA, nullptr, 2*D_IN, nullptr, nullptr, 0, MROWS, 2*D_IN, D_MODEL);
    // --- 3) causal conv + silu -> x_c (f32 + bf16) ---
    conv_silu_kernel<<<cdiv(NTOK,256), blk, 0, stream>>>(bufA, conv_w, conv_b, bufB, bufE, SEQ, (int)NTOK);
    // --- 4) x_dbl = x_c @ x_proj^T (padded N=128) ---
    gemm_bf16_kernel<<<dim3(XDBL_LD/128, MROWS/64), blk, 0, stream>>>(
        bufE, D_IN, wXpB, D_IN, xdblF, xdblB, XDBL_LD, nullptr, nullptr, 0, MROWS, XDBL_LD, D_IN);
    // --- 5) delta = softplus(dt @ dt_proj^T + b) ---
    gemm_bf16_kernel<<<dim3(D_IN/128, MROWS/64), blk, 0, stream>>>(
        xdblB, XDBL_LD, wDtB, DT_RANK, bufC, nullptr, D_IN, dt_proj_b, nullptr, 1, MROWS, D_IN, DT_RANK);
    // --- 6) selective scan -> y ---
    scan_kernel<<<(BATCH*D_IN)/256, blk, 0, stream>>>(bufC, xdblF, bufB, A_log, bufD, SEQ);
    // --- 7) y2 = (y + x_c*Dp) * silu(z)  (bf16, reuses bufE) ---
    gate_kernel<<<cdiv(NTOK,256), blk, 0, stream>>>(bufD, bufB, bufA, Dp, bufE, (int)NTOK);
    // --- 8) h = y2 @ out_proj^T  (reuses bufB) ---
    gemm_bf16_kernel<<<dim3(D_MODEL/128, MROWS/64), blk, 0, stream>>>(
        bufE, D_IN, wOutB, D_IN, bufB, nullptr, D_MODEL, nullptr, nullptr, 0, MROWS, D_MODEL, D_IN);
    // --- 9) src2 = src + ln(h; mnorm)  (reuses bufD) ---
    ln_kernel<<<MROWS, blk, 0, stream>>>(bufB, mnorm_g, mnorm_b, nullptr, bufD, src, D_MODEL);
    // --- 10) ln2(src2) -> bf16 ---
    ln_kernel<<<MROWS, blk, 0, stream>>>(bufD, n2_g, n2_b, lnB, nullptr, nullptr, D_MODEL);
    // --- 11) f = leaky(ln2 @ ffn_w1^T) -> bf16 (reuses bufA) ---
    gemm_bf16_kernel<<<dim3(D_FFN/128, MROWS/64), blk, 0, stream>>>(
        lnB, D_MODEL, wF1B, D_MODEL, nullptr, (bf16_t*)bufA, D_FFN, nullptr, nullptr, 2, MROWS, D_FFN, D_MODEL);
    // --- 12) src3 = src2 + f @ ffn_w2^T  (reuses bufC) ---
    gemm_bf16_kernel<<<dim3(D_MODEL/128, MROWS/64), blk, 0, stream>>>(
        (bf16_t*)bufA, D_FFN, wF2B, D_FFN, bufC, nullptr, D_MODEL, nullptr, bufD, 0, MROWS, D_MODEL, D_FFN);
    // --- 13) ln3(src3) -> bf16 ---
    ln_kernel<<<MROWS, blk, 0, stream>>>(bufC, n3_g, n3_b, lnB, nullptr, nullptr, D_MODEL);
    // --- 14) out = src3 + ln3 @ fe_w^T + fe_b ---
    gemm_bf16_kernel<<<dim3(D_MODEL/128, MROWS/64), blk, 0, stream>>>(
        lnB, D_MODEL, wFeB, D_MODEL, (float*)d_out, nullptr, D_MODEL, fe_b, bufC, 0, MROWS, D_MODEL, D_MODEL);
}
